// Involution_55525337203290
// MI455X (gfx1250) — matrix operand order repaired and checked by reading, not compiled, not run on hardware
//
#include <hip/hip_runtime.h>

// Problem constants (from reference)
#define BATCH  8
#define CIN    256
#define CR     64          // reduced channels
#define HH     112
#define WW     112
#define HW     (HH * WW)   // 12544 (multiple of 16)
#define GROUPS 16
#define KK     7
#define KK2    49
#define KO     (KK2 * GROUPS)  // 784 = 49 * 16  -> 49 WMMA M-tiles
#define PADR   3
#define EPSV   1e-5f

typedef __attribute__((ext_vector_type(2)))  float    v2f;
typedef __attribute__((ext_vector_type(8)))  float    v8f;
typedef __attribute__((ext_vector_type(16))) _Float16 v16h;

// ---------------------------------------------------------------------------
// Kernel 0: zero the BN-stats accumulator (d_ws is poisoned to 0xAA)
// ---------------------------------------------------------------------------
__global__ void k_zero(float* __restrict__ p) {
    p[threadIdx.x] = 0.0f;
}

// ---------------------------------------------------------------------------
// Kernel 1: t = x (B,C,H,W) . w1 (Cr,C) + b1 via V_WMMA_F32_16X16X32_F16
// (f16 inputs, f32 accumulate: K=256 reduction, ~1e-3 rel err, BN-safe).
// Each block: 8 waves covering 4 M-tiles (Cr=64) x 2 N-tiles (32 pixels).
// Also accumulates per-channel sum / sum-of-squares for BatchNorm.
// ---------------------------------------------------------------------------
__global__ void __launch_bounds__(256) k_gemm1(const float* __restrict__ x,
                                               const float* __restrict__ w1,
                                               const float* __restrict__ b1,
                                               float* __restrict__ t,
                                               float* __restrict__ stats) {
    __shared__ float s_sum[CR];
    __shared__ float s_sq[CR];
    const int tid = threadIdx.x;
    if (tid < CR) { s_sum[tid] = 0.0f; s_sq[tid] = 0.0f; }
    __syncthreads();

    const int wave = tid >> 5;
    const int lane = tid & 31;
    const int half = lane >> 4;   // lane-half selects K sub-block (ISA 7.12.2)
    const int l16  = lane & 15;   // M (for A) / N (for B,D) index

    const int mt    = wave & 3;                       // M-tile: channels mt*16..+15
    const int ntile = blockIdx.x * 2 + (wave >> 2);   // 16-pixel tile

    const int n0 = ntile * 16;          // HW % 16 == 0 -> tile never crosses batch
    const int b  = n0 / HW;
    const int hw = (n0 % HW) + l16;     // this lane's pixel (N index)

    const float* xb   = x  + (size_t)b * CIN * HW + hw;
    const float* arow = w1 + (size_t)(mt * 16 + l16) * CIN;   // A: row M=l16

    v8f acc = {};
    for (int k0 = 0; k0 < CIN; k0 += 32) {
        v16h a, bb;
        #pragma unroll
        for (int i = 0; i < 16; ++i) {
            // A 16x32 f16, k order chosen here (same as B): lanes<16 K=0..15, lanes>=16 K=16..31 ; M=l16
            const int kA = k0 + 16 * half + i;
            a[i] = (_Float16)arow[kA];
            // B 32x16 f16, k order chosen here (same as A): lanes<16 K=0..15, lanes>=16 K=16..31 ; N=l16
            const int kB = k0 + 16 * half + i;
            bb[i] = (_Float16)xb[(size_t)kB * HW];
        }
        acc = __builtin_amdgcn_wmma_f32_16x16x32_f16(
                  false, a, false, bb, (short)0, acc, false, false);
    }

    // D layout: vgpr r, lanes<16 -> M=r ; lanes>=16 -> M=r+8 ; N=l16
    float* trow = t + (size_t)b * CR * HW + hw;
    #pragma unroll
    for (int r = 0; r < 8; ++r) {
        const int m = mt * 16 + r + 8 * half;
        const float v = acc[r] + b1[m];
        trow[(size_t)m * HW] = v;

        // reduce across the 16 lanes sharing channel m
        float s = v, q = v * v;
        #pragma unroll
        for (int off = 8; off >= 1; off >>= 1) {
            s += __shfl_xor(s, off, 16);
            q += __shfl_xor(q, off, 16);
        }
        if (l16 == 0) {
            atomicAdd(&s_sum[m], s);   // ds_add_f32
            atomicAdd(&s_sq[m],  q);
        }
    }
    __syncthreads();
    if (tid < CR) {
        atomicAdd(&stats[tid],      s_sum[tid]);   // global_atomic_add_f32
        atomicAdd(&stats[CR + tid], s_sq[tid]);
    }
}

// ---------------------------------------------------------------------------
// Kernel 2: fold BN statistics + gamma/beta into per-channel scale/shift
// ---------------------------------------------------------------------------
__global__ void k_finalize(const float* __restrict__ stats,
                           const float* __restrict__ gamma,
                           const float* __restrict__ beta,
                           float* __restrict__ ss) {
    const int m = threadIdx.x;
    if (m < CR) {
        const float invN = 1.0f / (float)((size_t)BATCH * HW);
        const float mean = stats[m] * invN;
        const float var  = stats[CR + m] * invN - mean * mean;
        const float sc   = gamma[m] * rsqrtf(var + EPSV);
        ss[m]      = sc;
        ss[CR + m] = beta[m] - mean * sc;
    }
}

// ---------------------------------------------------------------------------
// Kernel 3 (fused): per 16-pixel tile
//   1) async-copy raw t tile (64 x 16) into LDS (GLOBAL_LOAD_ASYNC_TO_LDS_B32,
//      ASYNCcnt), then BN+ReLU in place
//   2) WMMA f16: kernel = w2 (784x64) . tn (64x16) + b2 -> 784x16 tile in LDS
//   3) involution: thread = channel c; 7x7 dynamic kernel vs padded x
// Avoids ever writing the 315 MB kernel tensor to HBM.
// ---------------------------------------------------------------------------
__global__ void __launch_bounds__(256) k_fused(const float* __restrict__ x,
                                               const float* __restrict__ t,
                                               const float* __restrict__ ss,
                                               const float* __restrict__ w2,
                                               const float* __restrict__ b2,
                                               float* __restrict__ out) {
    __shared__ float tn[CR * 16];   // 4 KB normalized-t tile
    __shared__ float kl[KO * 16];   // 49 KB kernel tile (784 rows x 16 pixels)

    const int tid   = threadIdx.x;
    const int ntile = blockIdx.x;
    const int n0    = ntile * 16;
    const int b     = n0 / HW;
    const int hw0   = n0 % HW;

    // ---- stage raw t tile via CDNA5 async global->LDS copies ----
    const float* tsrc = t + (size_t)b * CR * HW + hw0;
    #pragma unroll
    for (int r = 0; r < 4; ++r) {
        const int idx = tid + r * 256;           // 0..1023 = 64 ch x 16 px
        const int ch  = idx >> 4, p = idx & 15;
        const float* src = tsrc + (size_t)ch * HW + p;
        __attribute__((address_space(3))) float* dst =
            (__attribute__((address_space(3))) float*)&tn[idx];
        asm volatile("global_load_async_to_lds_b32 %0, %1, off"
                     :: "v"(dst), "v"(src) : "memory");
    }
    asm volatile("s_wait_asynccnt 0x0" ::: "memory");
    __syncthreads();
    // BN + ReLU in place (each thread touches only its own elements)
    #pragma unroll
    for (int r = 0; r < 4; ++r) {
        const int idx = tid + r * 256;
        const int ch  = idx >> 4;
        tn[idx] = fmaxf(tn[idx] * ss[ch] + ss[CR + ch], 0.0f);
    }
    __syncthreads();

    // ---- GEMM2 via WMMA f16: 49 M-tiles over 8 waves ----
    const int wave = tid >> 5;
    const int lane = tid & 31;
    const int half = lane >> 4;
    const int l16  = lane & 15;

    for (int mt = wave; mt < KO / 16; mt += 8) {   // wave-uniform: EXEC all-1s
        const float* arow = w2 + (size_t)(mt * 16 + l16) * CR;
        v8f acc = {};
        #pragma unroll
        for (int k0 = 0; k0 < CR; k0 += 32) {
            v16h a, bb;
            #pragma unroll
            for (int i = 0; i < 16; ++i) {
                const int kA = k0 + 16 * half + i;
                a[i] = (_Float16)arow[kA];
                const int kB = k0 + 16 * half + i;
                bb[i] = (_Float16)tn[kB * 16 + l16];
            }
            acc = __builtin_amdgcn_wmma_f32_16x16x32_f16(
                      false, a, false, bb, (short)0, acc, false, false);
        }
        #pragma unroll
        for (int r = 0; r < 8; ++r) {
            const int row = mt * 16 + r + 8 * half;
            kl[row * 16 + l16] = acc[r] + b2[row];
        }
    }
    __syncthreads();

    // ---- involution: one channel per thread, 16 pixels in registers ----
    const int c  = tid;          // 0..255
    const int g  = c >> 4;       // group
    const int h  = hw0 / WW;
    const int w0 = hw0 % WW;     // multiple of 16: tile stays in one row
    const float* xc = x + ((size_t)b * CIN + c) * HW;

    float accp[16];
    #pragma unroll
    for (int p = 0; p < 16; ++p) accp[p] = 0.0f;

    for (int i = 0; i < KK; ++i) {
        const int hh = h + i - PADR;
        if (hh < 0 || hh >= HH) continue;
        const float* xr = xc + (size_t)hh * WW;
        __builtin_prefetch(xr + w0, 0, 0);   // global_prefetch_b8
        float xrow[16 + KK - 1];
        #pragma unroll
        for (int q = 0; q < 16 + KK - 1; ++q) {
            const int ww = w0 - PADR + q;
            xrow[q] = (ww >= 0 && ww < WW) ? xr[ww] : 0.0f;
        }
        const float* krow = &kl[(g * KK2 + i * KK) * 16];
        #pragma unroll
        for (int j = 0; j < KK; ++j) {
            #pragma unroll
            for (int p = 0; p < 16; ++p)
                accp[p] += krow[j * 16 + p] * xrow[p + j];
        }
    }

    float* orow = out + ((size_t)b * CIN + c) * HW + hw0;
    #pragma unroll
    for (int p = 0; p < 16; ++p) orow[p] = accp[p];
}

// ---------------------------------------------------------------------------
// Host launcher
// ---------------------------------------------------------------------------
extern "C" void kernel_launch(void* const* d_in, const int* in_sizes, int n_in,
                              void* d_out, int out_size, void* d_ws, size_t ws_size,
                              hipStream_t stream) {
    const float* x     = (const float*)d_in[0];
    const float* w1    = (const float*)d_in[1];
    const float* b1    = (const float*)d_in[2];
    const float* gamma = (const float*)d_in[3];
    const float* beta  = (const float*)d_in[4];
    const float* w2    = (const float*)d_in[5];
    const float* b2    = (const float*)d_in[6];
    float* out = (float*)d_out;

    // workspace layout: t (B*CR*HW f32) | stats (128 f32) | scale/shift (128 f32)
    float* t     = (float*)d_ws;
    float* stats = t + (size_t)BATCH * CR * HW;
    float* ss    = stats + 2 * CR;

    const int ntiles = BATCH * HW / 16;   // 6272 16-pixel tiles

    hipLaunchKernelGGL(k_zero,     dim3(1),          dim3(2 * CR), 0, stream, stats);
    hipLaunchKernelGGL(k_gemm1,    dim3(ntiles / 2), dim3(256),    0, stream,
                       x, w1, b1, t, stats);
    hipLaunchKernelGGL(k_finalize, dim3(1),          dim3(CR),     0, stream,
                       stats, gamma, beta, ss);
    hipLaunchKernelGGL(k_fused,    dim3(ntiles),     dim3(256),    0, stream,
                       x, t, ss, w2, b2, out);
}
